// Model_62397284876313
// MI455X (gfx1250) — matrix operand order repaired and checked by reading, not compiled, not run on hardware
//
#include <hip/hip_runtime.h>
#include <hip/hip_bf16.h>
#include <math.h>

// ---------------- problem constants ----------------
constexpr int Bc  = 4;
constexpr int Qc  = 64;
constexpr int Hc  = 4096;
constexpr int NHc = 32;
constexpr int KVHc= 8;
constexpr int HDc = 128;
constexpr int Pc  = 4096;
constexpr int NKVc = Pc + Qc;          // 4160
constexpr int NQK  = NHc * HDc;        // 4096
constexpr int NKN  = KVHc * HDc;       // 1024
constexpr int NQKV = NQK + 2 * NKN;    // 6144
constexpr int Mrows = Bc * Qc;         // 256
constexpr float SCALE = 0.08838834764831845f; // 1/sqrt(128)

// ---------------- WMMA types ----------------
typedef __bf16 bf16_t;
typedef __attribute__((ext_vector_type(16))) __bf16 v16bf;
typedef __attribute__((ext_vector_type(8)))  float  v8f;

// native hardware convert (v_cvt_bf16_f32 / packed v_cvt_pk_bf16_f32)
static __device__ __forceinline__ bf16_t f2bf(float f) { return (bf16_t)f; }

// =====================================================================
// Kernel 1: fused QKV projection. C[256 x 6144] = hidden[256x4096] x W
// One wave computes one 16x16 tile, K-loop step 32, bf16 WMMA, f32 acc.
// =====================================================================
__global__ __launch_bounds__(128) void qkv_gemm_kernel(
    const float* __restrict__ hidden,
    const float* __restrict__ Wq, const float* __restrict__ Wk,
    const float* __restrict__ Wv,
    float* __restrict__ qkv_raw) {
  const int lane  = threadIdx.x & 31;
  const int wave  = threadIdx.x >> 5;
  const int tileM = blockIdx.x;                 // 0..15
  const int tileN = blockIdx.y * 4 + wave;      // 0..383
  const int n0 = tileN * 16;

  const float* Wsrc; int ld, col0;
  if (n0 < NQK)             { Wsrc = Wq; ld = NQK; col0 = n0; }
  else if (n0 < NQK + NKN)  { Wsrc = Wk; ld = NKN; col0 = n0 - NQK; }
  else                      { Wsrc = Wv; ld = NKN; col0 = n0 - NQK - NKN; }

  const int lm    = lane & 15;
  const int khalf = (lane >> 4) << 3;           // 0 or 8
  const int row   = tileM * 16 + lm;
  const int col   = col0 + lm;

  v8f acc = {};
  for (int k0 = 0; k0 < Hc; k0 += 32) {
    __builtin_prefetch(hidden + (size_t)row * Hc + k0 + 64, 0, 1);
    __builtin_prefetch(Wsrc + (size_t)(k0 + 32) * ld + col, 0, 1);
    v16bf a, bb;
#pragma unroll
    for (int v = 0; v < 8; ++v) {
      const int kk = k0 + ((v >> 2) << 4) + khalf + ((v & 3) << 1);
      float2 av = *reinterpret_cast<const float2*>(hidden + (size_t)row * Hc + kk);
      a[2 * v]     = f2bf(av.x);
      a[2 * v + 1] = f2bf(av.y);
      bb[2 * v]     = f2bf(Wsrc[(size_t)kk * ld + col]);
      bb[2 * v + 1] = f2bf(Wsrc[(size_t)(kk + 1) * ld + col]);
    }
    acc = __builtin_amdgcn_wmma_f32_16x16x32_bf16(false, a, false, bb,
                                                  (short)0, acc, false, false);
  }
#pragma unroll
  for (int r = 0; r < 8; ++r) {
    const int m = r + khalf;
    qkv_raw[(size_t)(tileM * 16 + m) * NQKV + n0 + lm] = acc[r];
  }
}

// =====================================================================
// Kernel 2: per-head LayerNorm + RoPE, emit bf16 q / new-k / new-v.
// One wave handles one 128-element head row (4 elems/lane, stride 32).
// =====================================================================
__global__ __launch_bounds__(128) void ln_rope_kernel(
    const float* __restrict__ qkv_raw,
    const float* __restrict__ q_ln_w, const float* __restrict__ q_ln_b,
    const float* __restrict__ k_ln_w, const float* __restrict__ k_ln_b,
    const int*   __restrict__ position_ids,
    bf16_t* __restrict__ q_bf,
    bf16_t* __restrict__ k_bf,
    bf16_t* __restrict__ v_bf) {
  const int lane = threadIdx.x & 31;
  const int wave = threadIdx.x >> 5;
  const int row  = blockIdx.x * 4 + wave;       // 0..12287
  const int QROWS = Bc * Qc * NHc;              // 8192
  const int KROWS = Bc * Qc * KVHc;             // 2048

  int b, q, h, kind, srcCol, dstBase;
  const float *lnw = nullptr, *lnb = nullptr;
  bf16_t* dst;
  if (row < QROWS) {
    kind = 0; int t = row; h = t % NHc; t /= NHc; q = t % Qc; b = t / Qc;
    srcCol = h * HDc; lnw = q_ln_w + h * HDc; lnb = q_ln_b + h * HDc;
    dst = q_bf; dstBase = ((b * NHc + h) * Qc + q) * HDc;
  } else if (row < QROWS + KROWS) {
    kind = 1; int t = row - QROWS; h = t % KVHc; t /= KVHc; q = t % Qc; b = t / Qc;
    srcCol = NQK + h * HDc; lnw = k_ln_w + h * HDc; lnb = k_ln_b + h * HDc;
    dst = k_bf; dstBase = ((b * KVHc + h) * Qc + q) * HDc;
  } else {
    kind = 2; int t = row - QROWS - KROWS; h = t % KVHc; t /= KVHc; q = t % Qc; b = t / Qc;
    srcCol = NQK + NKN + h * HDc;
    dst = v_bf; dstBase = ((b * KVHc + h) * Qc + q) * HDc;
  }

  const float* src = qkv_raw + (size_t)(b * Qc + q) * NQKV + srcCol;
  float x[4];
#pragma unroll
  for (int i = 0; i < 4; ++i) x[i] = src[lane + 32 * i];

  if (kind < 2) {
    float s  = x[0] + x[1] + x[2] + x[3];
    float s2 = x[0]*x[0] + x[1]*x[1] + x[2]*x[2] + x[3]*x[3];
#pragma unroll
    for (int msk = 1; msk < 32; msk <<= 1) {
      s  += __shfl_xor(s,  msk, 32);
      s2 += __shfl_xor(s2, msk, 32);
    }
    const float mu  = s * (1.0f / HDc);
    const float var = s2 * (1.0f / HDc) - mu * mu;
    const float rs  = rsqrtf(var + 1e-5f);
#pragma unroll
    for (int i = 0; i < 4; ++i) {
      const int d = lane + 32 * i;
      x[i] = (x[i] - mu) * rs * lnw[d] + lnb[d];
    }
    const int pos = position_ids[b * Qc + q];
#pragma unroll
    for (int i = 0; i < 2; ++i) {
      const int d = lane + 32 * i;                       // 0..63
      const float invf = __powf(10000.0f, -(float)(2 * d) * (1.0f / HDc));
      const float th = (float)pos * invf;
      float sn, cs; __sincosf(th, &sn, &cs);
      const float lo = x[i], hi = x[i + 2];
      x[i]     = lo * cs - hi * sn;
      x[i + 2] = hi * cs + lo * sn;
    }
  }
#pragma unroll
  for (int i = 0; i < 4; ++i) dst[dstBase + lane + 32 * i] = f2bf(x[i]);
}

// =====================================================================
// Kernel 3: flash attention over 4160 keys. Block = 4 waves = 4 q-tiles
// of one (b, head). Scores: 4xWMMA (K=128). Probs transposed via LDS.
// Context: 8xWMMA per 32-key chunk.
// =====================================================================
__global__ __launch_bounds__(128) void attn_kernel(
    const bf16_t* __restrict__ q_bf,
    const bf16_t* __restrict__ k_bf,
    const bf16_t* __restrict__ v_bf,
    const float* __restrict__ past_k,
    const float* __restrict__ past_v,
    const float* __restrict__ attn_mask,
    bf16_t* __restrict__ ctx_bf) {
  __shared__ bf16_t ldsP[4][16 * 32];
  const int lane = threadIdx.x & 31;
  const int wave = threadIdx.x >> 5;
  const int bh = blockIdx.x;
  const int b = bh / NHc, h = bh % NHc;
  const int kvh = h / (NHc / KVHc);
  const int q0 = wave * 16;
  const int lm = lane & 15;
  const int khalf = (lane >> 4) << 3;

  // Q fragments (16 rows x 128 dims -> 4 A fragments of 16x32)
  v16bf qa[4];
  const bf16_t* qrow = q_bf + ((size_t)(b * NHc + h) * Qc + q0 + lm) * HDc;
#pragma unroll
  for (int t = 0; t < 4; ++t) {
#pragma unroll
    for (int v = 0; v < 8; ++v) {
      const int d = 32 * t + ((v >> 2) << 4) + khalf + ((v & 3) << 1);
      qa[t][2 * v]     = qrow[d];
      qa[t][2 * v + 1] = qrow[d + 1];
    }
  }

  v8f O[8];
#pragma unroll
  for (int t = 0; t < 8; ++t) O[t] = (v8f){};
  float mrow[8], srow[8];
#pragma unroll
  for (int r = 0; r < 8; ++r) { mrow[r] = -INFINITY; srow[r] = 0.0f; }

  const float* pk = past_k + (size_t)(b * KVHc + kvh) * Pc * HDc;
  const float* pv = past_v + (size_t)(b * KVHc + kvh) * Pc * HDc;
  const bf16_t* nk = k_bf + (size_t)(b * KVHc + kvh) * Qc * HDc;
  const bf16_t* nv = v_bf + (size_t)(b * KVHc + kvh) * Qc * HDc;
  const float* mask = attn_mask + (size_t)b * Qc * NKVc;

  for (int c0 = 0; c0 < NKVc; c0 += 32) {
    // prefetch next chunk of K and V (gfx1250 global_prefetch path)
    if (c0 + 32 < Pc) {
      __builtin_prefetch(pk + (size_t)(c0 + 32 + (lane & 15)) * HDc + khalf * 2, 0, 1);
      __builtin_prefetch(pv + (size_t)(c0 + 32 + (lane & 15)) * HDc + khalf * 2, 0, 1);
    }
    v8f st[2];
#pragma unroll
    for (int sub = 0; sub < 2; ++sub) {
      const int kvbase = c0 + sub * 16;
      const int kv = kvbase + lm;
      v16bf kb[4];
#pragma unroll
      for (int t = 0; t < 4; ++t) {
      if (kv < Pc) {
        const float* kr = pk + (size_t)kv * HDc;
#pragma unroll
        for (int v = 0; v < 8; ++v) {
          const int d = 32 * t + ((v >> 2) << 4) + khalf + ((v & 3) << 1);
          float2 f = *reinterpret_cast<const float2*>(kr + d);
          kb[t][2 * v]     = f2bf(f.x);
          kb[t][2 * v + 1] = f2bf(f.y);
        }
      } else {
        const bf16_t* kr = nk + (size_t)(kv - Pc) * HDc;
#pragma unroll
        for (int v = 0; v < 8; ++v) {
          const int d = 32 * t + ((v >> 2) << 4) + khalf + ((v & 3) << 1);
          kb[t][2 * v]     = kr[d];
          kb[t][2 * v + 1] = kr[d + 1];
        }
      }
      }
      v8f s = {};
#pragma unroll
      for (int t = 0; t < 4; ++t)
        s = __builtin_amdgcn_wmma_f32_16x16x32_bf16(false, qa[t], false, kb[t],
                                                    (short)0, s, false, false);
#pragma unroll
      for (int r = 0; r < 8; ++r) s[r] = s[r] * SCALE;
      if (kvbase >= Pc) {                         // tree-mask region only
#pragma unroll
        for (int r = 0; r < 8; ++r) {
          const int qi = q0 + r + khalf;
          s[r] = s[r] + mask[(size_t)qi * NKVc + kvbase + lm];
        }
      }
      st[sub] = s;
    }

    // online softmax update over the 32 new columns
#pragma unroll
    for (int r = 0; r < 8; ++r) {
      float mx = fmaxf(st[0][r], st[1][r]);
#pragma unroll
      for (int msk = 1; msk < 16; msk <<= 1) mx = fmaxf(mx, __shfl_xor(mx, msk, 32));
      const float mnew = fmaxf(mrow[r], mx);
      const float fr = __expf(mrow[r] - mnew);
      const float p0 = __expf(st[0][r] - mnew);
      const float p1 = __expf(st[1][r] - mnew);
      float ps = p0 + p1;
#pragma unroll
      for (int msk = 1; msk < 16; msk <<= 1) ps += __shfl_xor(ps, msk, 32);
      srow[r] = srow[r] * fr + ps;
      mrow[r] = mnew;
#pragma unroll
      for (int t = 0; t < 8; ++t) O[t][r] = O[t][r] * fr;
      const int mi = r + khalf;                  // prob row in 16x32 tile
      ldsP[wave][mi * 32 + lm]      = f2bf(p0);
      ldsP[wave][mi * 32 + 16 + lm] = f2bf(p1);
    }
    asm volatile("s_wait_dscnt 0" ::: "memory");

    // transpose probs into A layout
    v16bf pa;
#pragma unroll
    for (int v = 0; v < 8; ++v) {
      const int n = ((v >> 2) << 4) + khalf + ((v & 3) << 1);
      pa[2 * v]     = ldsP[wave][lm * 32 + n];
      pa[2 * v + 1] = ldsP[wave][lm * 32 + n + 1];
    }

    // context accumulation: 8 head-dim tiles of 16
#pragma unroll
    for (int t = 0; t < 8; ++t) {
      v16bf vb;
      const int dcol = t * 16 + lm;
#pragma unroll
      for (int v = 0; v < 8; ++v) {
        const int kidx = ((v >> 2) << 4) + khalf + ((v & 3) << 1);
        const int kv = c0 + kidx;
        if (kv < Pc) {
          vb[2 * v]     = f2bf(pv[(size_t)kv * HDc + dcol]);
          vb[2 * v + 1] = f2bf(pv[(size_t)(kv + 1) * HDc + dcol]);
        } else {
          vb[2 * v]     = nv[(size_t)(kv - Pc) * HDc + dcol];
          vb[2 * v + 1] = nv[(size_t)(kv + 1 - Pc) * HDc + dcol];
        }
      }
      O[t] = __builtin_amdgcn_wmma_f32_16x16x32_bf16(false, pa, false, vb,
                                                     (short)0, O[t], false, false);
    }
  }

  // epilogue: normalize + store bf16 ctx in [B, Q, NH*HD]
#pragma unroll
  for (int t = 0; t < 8; ++t) {
#pragma unroll
    for (int r = 0; r < 8; ++r) {
      const int qi = q0 + r + khalf;
      const int d = t * 16 + lm;
      const float val = O[t][r] / srow[r];
      ctx_bf[(size_t)(b * Qc + qi) * NQK + h * HDc + d] = f2bf(val);
    }
  }
}

// =====================================================================
// Kernel 4: output projection. out[256x4096] = ctx_bf16[256x4096] x Wo
// =====================================================================
__global__ __launch_bounds__(128) void outproj_kernel(
    const bf16_t* __restrict__ ctx_bf,
    const float* __restrict__ Wo,
    float* __restrict__ out) {
  const int lane  = threadIdx.x & 31;
  const int wave  = threadIdx.x >> 5;
  const int tileM = blockIdx.x;
  const int tileN = blockIdx.y * 4 + wave;
  const int n0 = tileN * 16;
  const int lm = lane & 15;
  const int khalf = (lane >> 4) << 3;
  const int row = tileM * 16 + lm;
  const int col = n0 + lm;

  v8f acc = {};
  for (int k0 = 0; k0 < NQK; k0 += 32) {
    __builtin_prefetch(Wo + (size_t)(k0 + 32) * Hc + col, 0, 1);
    v16bf a, bb;
#pragma unroll
    for (int v = 0; v < 8; ++v) {
      const int kk = k0 + ((v >> 2) << 4) + khalf + ((v & 3) << 1);
      a[2 * v]     = ctx_bf[(size_t)row * NQK + kk];
      a[2 * v + 1] = ctx_bf[(size_t)row * NQK + kk + 1];
      bb[2 * v]     = f2bf(Wo[(size_t)kk * Hc + col]);
      bb[2 * v + 1] = f2bf(Wo[(size_t)(kk + 1) * Hc + col]);
    }
    acc = __builtin_amdgcn_wmma_f32_16x16x32_bf16(false, a, false, bb,
                                                  (short)0, acc, false, false);
  }
#pragma unroll
  for (int r = 0; r < 8; ++r) {
    const int m = r + khalf;
    out[(size_t)(tileM * 16 + m) * Hc + n0 + lm] = acc[r];
  }
}

// =====================================================================
extern "C" void kernel_launch(void* const* d_in, const int* in_sizes, int n_in,
                              void* d_out, int out_size, void* d_ws, size_t ws_size,
                              hipStream_t stream) {
  const float* hidden = (const float*)d_in[0];
  const float* Wq     = (const float*)d_in[1];
  const float* Wk     = (const float*)d_in[2];
  const float* Wv     = (const float*)d_in[3];
  const float* Wo     = (const float*)d_in[4];
  const float* q_ln_w = (const float*)d_in[5];
  const float* q_ln_b = (const float*)d_in[6];
  const float* k_ln_w = (const float*)d_in[7];
  const float* k_ln_b = (const float*)d_in[8];
  const float* past_k = (const float*)d_in[9];
  const float* past_v = (const float*)d_in[10];
  const float* amask  = (const float*)d_in[11];
  const int*   posids = (const int*)d_in[12];

  // workspace carve-out
  char* ws = (char*)d_ws;
  const size_t qkv_bytes  = (size_t)Mrows * NQKV * sizeof(float);        // 6 MB
  const size_t q_bytes    = (size_t)Bc * NHc  * Qc * HDc * 2;            // 2 MB
  const size_t kv_bytes   = (size_t)Bc * KVHc * Qc * HDc * 2;            // 0.5 MB
  float*  qkv_raw = (float*)ws;
  bf16_t* q_bfp   = (bf16_t*)(ws + qkv_bytes);
  bf16_t* k_bfp   = (bf16_t*)(ws + qkv_bytes + q_bytes);
  bf16_t* v_bfp   = (bf16_t*)(ws + qkv_bytes + q_bytes + kv_bytes);
  bf16_t* ctx_bfp = (bf16_t*)(ws + qkv_bytes + q_bytes + 2 * kv_bytes);

  qkv_gemm_kernel<<<dim3(Mrows / 16, NQKV / 16 / 4), 128, 0, stream>>>(
      hidden, Wq, Wk, Wv, qkv_raw);

  const int rows = Bc * Qc * (NHc + 2 * KVHc);   // 12288
  ln_rope_kernel<<<dim3(rows / 4), 128, 0, stream>>>(
      qkv_raw, q_ln_w, q_ln_b, k_ln_w, k_ln_b, posids, q_bfp, k_bfp, v_bfp);

  attn_kernel<<<dim3(Bc * NHc), 128, 0, stream>>>(
      q_bfp, k_bfp, v_bfp, past_k, past_v, amask, ctx_bfp);

  outproj_kernel<<<dim3(Mrows / 16, Hc / 16 / 4), 128, 0, stream>>>(
      ctx_bfp, Wo, (float*)d_out);
}
